// VisionZipCompressor_28278064677485
// MI455X (gfx1250) — matrix operand order repaired and checked by reading, not compiled, not run on hardware
//
#include <hip/hip_runtime.h>
#include <hip/hip_bf16.h>

typedef __attribute__((ext_vector_type(16))) __bf16 v16bf;
typedef __attribute__((ext_vector_type(8)))  float  v8f;

#define B_   8
#define H_   16
#define L_   577
#define N_   576
#define D_   1024
#define NEG_INF (-3.0e38f)
#define EPS_ 1e-12f

union Frag { v16bf v; uint4 q[2]; };

__device__ __forceinline__ unsigned short f2bf_rne(float f) {
    unsigned int u = __float_as_uint(f);
    unsigned int r = u + 0x7fffu + ((u >> 16) & 1u);
    return (unsigned short)(r >> 16);
}

// ---------------------------------------------------------------------------
// Kernel 1: s_attn[b,n] = mean over H of attn[b,h,0,1+n]
// ---------------------------------------------------------------------------
__global__ __launch_bounds__(256) void attn_mean_kernel(const float* __restrict__ attn,
                                                        float* __restrict__ sattn) {
    int idx = blockIdx.x * 256 + threadIdx.x;
    if (idx >= B_ * N_) return;
    int b = idx / N_, n = idx % N_;
    const float* base = attn + (size_t)b * H_ * L_ * L_ + (size_t)(1 + n);
    float s = 0.f;
    #pragma unroll
    for (int h = 0; h < H_; ++h)
        s += base[(size_t)h * L_ * L_];
    sattn[idx] = s * (1.0f / H_);
}

// ---------------------------------------------------------------------------
// Kernel 2: per patch-token row: feature softmax-entropy (tau=0.2) and
// unit-normalized row emitted as bf16 into z[b, n, :].
// ---------------------------------------------------------------------------
__global__ __launch_bounds__(256) void prep_kernel(const float* __restrict__ keys,
                                                   unsigned short* __restrict__ zb,
                                                   float* __restrict__ Hfeat) {
    int b = blockIdx.x / N_, n = blockIdx.x % N_;
    int tid = threadIdx.x;
    const float* x = keys + ((size_t)(b * L_ + 1 + n)) * D_;
    __shared__ float red[256];

    float v[4];
    float lmax = NEG_INF, lsq = 0.f;
    #pragma unroll
    for (int i = 0; i < 4; ++i) {
        v[i] = x[tid + i * 256];
        lsq += v[i] * v[i];
        lmax = fmaxf(lmax, v[i] * 5.0f);     // y = x / 0.2
    }
    // reduce max
    red[tid] = lmax; __syncthreads();
    for (int s = 128; s > 0; s >>= 1) { if (tid < s) red[tid] = fmaxf(red[tid], red[tid + s]); __syncthreads(); }
    float gmax = red[0]; __syncthreads();
    // reduce sum of squares
    red[tid] = lsq; __syncthreads();
    for (int s = 128; s > 0; s >>= 1) { if (tid < s) red[tid] += red[tid + s]; __syncthreads(); }
    float gsq = red[0]; __syncthreads();

    float ls = 0.f, lt = 0.f;
    #pragma unroll
    for (int i = 0; i < 4; ++i) {
        float y = v[i] * 5.0f;
        float e = __expf(y - gmax);
        ls += e; lt += y * e;
    }
    red[tid] = ls; __syncthreads();
    for (int s = 128; s > 0; s >>= 1) { if (tid < s) red[tid] += red[tid + s]; __syncthreads(); }
    float gs = red[0]; __syncthreads();
    red[tid] = lt; __syncthreads();
    for (int s = 128; s > 0; s >>= 1) { if (tid < s) red[tid] += red[tid + s]; __syncthreads(); }
    float gt = red[0]; __syncthreads();

    if (tid == 0) {
        // H = (logsumexp(y) - E_p[y]) / log(D)  (exact softmax entropy)
        float Hent = gmax + __logf(gs) - gt / gs;
        Hfeat[b * N_ + n] = Hent / __logf((float)D_);
    }
    float inv = 1.0f / (sqrtf(gsq) + EPS_);
    unsigned short* zr = zb + ((size_t)(b * N_ + n)) * D_;
    #pragma unroll
    for (int i = 0; i < 4; ++i)
        zr[tid + i * 256] = f2bf_rne(v[i] * inv);
}

// ---------------------------------------------------------------------------
// Kernel 3: sim = z z^T (bf16 WMMA, f32 accum), fused online softmax-entropy
// per row with diagonal masked; emits I = 1 - H/ln(576).
// One workgroup (4 waves) per 16-row strip of one batch; A strip in LDS
// (compiler hoists it to VGPRs), B software-pipelined from global/L2.
// ---------------------------------------------------------------------------
__global__ __launch_bounds__(128) void sim_entropy_kernel(const unsigned short* __restrict__ zb,
                                                          float* __restrict__ Isim) {
    const int strip = blockIdx.x % (N_ / 16);
    const int b     = blockIdx.x / (N_ / 16);
    const int r0    = strip * 16;

    __shared__ unsigned short As[16 * D_];          // 32 KB A strip
    __shared__ float pm[4][16], ps[4][16], pt[4][16];

    // cooperative A-strip load (contiguous 16*1024 bf16)
    {
        const uint4* src = (const uint4*)(zb + ((size_t)(b * N_ + r0)) * D_);
        uint4* dst = (uint4*)As;
        for (int i = threadIdx.x; i < 16 * D_ / 8; i += 128) dst[i] = src[i];
    }
    __syncthreads();

    const int lane = threadIdx.x & 31;
    const int wave = threadIdx.x >> 5;
    const int half = (lane & 16) ? 1 : 0;          // K-half / row-half selector

    float m[8], s[8], t[8];
    #pragma unroll
    for (int r = 0; r < 8; ++r) { m[r] = NEG_INF; s[r] = 0.f; t[r] = 0.f; }

    const unsigned short* aR = As + (lane & 15) * D_ + half * 16;

    for (int jt = wave; jt < N_ / 16; jt += 4) {    // 9 column tiles per wave
        const int col = jt * 16 + (lane & 15);
        v8f c = {};
        const uint4* pB = (const uint4*)(zb + ((size_t)(b * N_ + col)) * D_ + half * 16);
        // software pipeline: prefetch next K-chunk's B while WMMA consumes current
        uint4 nb0 = pB[0], nb1 = pB[1];
        #pragma unroll
        for (int kk = 0; kk < D_; kk += 32) {
            Frag a, bv;
            a.q[0]  = *(const uint4*)(aR + kk);
            a.q[1]  = *(const uint4*)(aR + kk + 8);
            bv.q[0] = nb0;
            bv.q[1] = nb1;
            if (kk + 32 < D_) {
                nb0 = pB[(kk >> 3) + 4];
                nb1 = pB[(kk >> 3) + 5];
            }
            c = __builtin_amdgcn_wmma_f32_16x16x32_bf16(
                    false, a.v, false, bv.v, (short)0, c, false, false);
        }
        // online (m, s, t) update; C layout: VGPR r -> row r (+8 for lanes>=16)
        #pragma unroll
        for (int r = 0; r < 8; ++r) {
            int rg = r0 + r + half * 8;
            if (rg == col) continue;                // diagonal masked (-1e9 -> exp=0)
            float y  = c[r] * 10.0f;                // / TAU_SIM
            float nm = fmaxf(m[r], y);
            float eo = __expf(m[r] - nm), en = __expf(y - nm);
            s[r] = s[r] * eo + en;
            t[r] = t[r] * eo + y * en;
            m[r] = nm;
        }
    }

    // cross-lane combine over the 16 lanes sharing each row (xor stays in half)
    #pragma unroll
    for (int mask = 1; mask < 16; mask <<= 1) {
        #pragma unroll
        for (int r = 0; r < 8; ++r) {
            float m2 = __shfl_xor(m[r], mask, 32);
            float s2 = __shfl_xor(s[r], mask, 32);
            float t2 = __shfl_xor(t[r], mask, 32);
            float nm = fmaxf(m[r], m2);
            float e1 = __expf(m[r] - nm), e2 = __expf(m2 - nm);
            s[r] = s[r] * e1 + s2 * e2;
            t[r] = t[r] * e1 + t2 * e2;
            m[r] = nm;
        }
    }
    if ((lane & 15) == 0) {
        #pragma unroll
        for (int r = 0; r < 8; ++r) {
            int row = r + half * 8;
            pm[wave][row] = m[r]; ps[wave][row] = s[r]; pt[wave][row] = t[r];
        }
    }
    __syncthreads();

    if (threadIdx.x < 16) {                         // cross-wave combine + entropy
        int row = threadIdx.x;
        float M = pm[0][row], S = ps[0][row], T = pt[0][row];
        #pragma unroll
        for (int w = 1; w < 4; ++w) {
            float m2 = pm[w][row], s2 = ps[w][row], t2 = pt[w][row];
            float nm = fmaxf(M, m2);
            float e1 = __expf(M - nm), e2 = __expf(m2 - nm);
            S = S * e1 + s2 * e2; T = T * e1 + t2 * e2; M = nm;
        }
        float Hs = M + __logf(S) - T / S;           // softmax entropy (nats)
        Isim[b * N_ + r0 + row] = 1.0f - Hs / __logf((float)N_);
    }
}

// ---------------------------------------------------------------------------
// Kernel 4: per-batch z-scores (ddof=1) combined: 1.0*sa + 0.4*H + 0.6*I
// ---------------------------------------------------------------------------
__global__ __launch_bounds__(256) void score_kernel(const float* __restrict__ sa,
                                                    const float* __restrict__ Hf,
                                                    const float* __restrict__ Is,
                                                    float* __restrict__ scores) {
    int b = blockIdx.x, tid = threadIdx.x;
    __shared__ float red[256];
    float sum[3] = {0.f, 0.f, 0.f}, sq[3] = {0.f, 0.f, 0.f};
    for (int n = tid; n < N_; n += 256) {
        float a = sa[b * N_ + n], h = Hf[b * N_ + n], i = Is[b * N_ + n];
        sum[0] += a; sq[0] += a * a;
        sum[1] += h; sq[1] += h * h;
        sum[2] += i; sq[2] += i * i;
    }
    float mean[3], sd[3];
    for (int c = 0; c < 3; ++c) {
        red[tid] = sum[c]; __syncthreads();
        for (int s = 128; s > 0; s >>= 1) { if (tid < s) red[tid] += red[tid + s]; __syncthreads(); }
        float S = red[0]; __syncthreads();
        red[tid] = sq[c]; __syncthreads();
        for (int s = 128; s > 0; s >>= 1) { if (tid < s) red[tid] += red[tid + s]; __syncthreads(); }
        float Q = red[0]; __syncthreads();
        mean[c] = S / (float)N_;
        sd[c] = sqrtf(fmaxf((Q - (float)N_ * mean[c] * mean[c]) / (float)(N_ - 1), 0.f));
    }
    for (int n = tid; n < N_; n += 256) {
        float a = sa[b * N_ + n], h = Hf[b * N_ + n], i = Is[b * N_ + n];
        scores[b * N_ + n] = 1.0f * (a - mean[0]) / (sd[0] + EPS_)
                           + 0.4f * (h - mean[1]) / (sd[1] + EPS_)
                           + 0.6f * (i - mean[2]) / (sd[2] + EPS_);
    }
}

// ---------------------------------------------------------------------------
// Kernel 5: top-k (iterative argmax, tie -> smallest index) + ascending keep[]
// ---------------------------------------------------------------------------
__global__ __launch_bounds__(256) void topk_kernel(const float* __restrict__ scores,
                                                   const int* __restrict__ dom,
                                                   int* __restrict__ keep) {
    int b = blockIdx.x, tid = threadIdx.x;
    __shared__ float sc[N_];
    __shared__ float rv[256];
    __shared__ int   ri[256];
    __shared__ int   sel[64];
    __shared__ int   kv[65];

    for (int n = tid; n < N_; n += 256) sc[n] = scores[b * N_ + n];
    __syncthreads();

    int k = dom[0]; k = k < 4 ? 4 : (k > 64 ? 64 : k);

    for (int it = 0; it < k; ++it) {
        float bv = NEG_INF; int bi = 1 << 30;
        for (int n = tid; n < N_; n += 256)
            if (sc[n] > bv || (sc[n] == bv && n < bi)) { bv = sc[n]; bi = n; }
        rv[tid] = bv; ri[tid] = bi; __syncthreads();
        for (int s = 128; s > 0; s >>= 1) {
            if (tid < s) {
                if (rv[tid + s] > rv[tid] ||
                    (rv[tid + s] == rv[tid] && ri[tid + s] < ri[tid])) {
                    rv[tid] = rv[tid + s]; ri[tid] = ri[tid + s];
                }
            }
            __syncthreads();
        }
        if (tid == 0) { sel[it] = ri[0]; sc[ri[0]] = NEG_INF; }
        __syncthreads();
    }
    if (tid <= k) kv[tid] = (tid == 0) ? 0 : sel[tid - 1] + 1;
    __syncthreads();
    if (tid <= k) {
        int v = kv[tid], rank = 0;
        for (int j = 0; j <= k; ++j) rank += (kv[j] < v);
        keep[b * 65 + rank] = v;
    }
}

// ---------------------------------------------------------------------------
// Kernel 6: gather dominant rows of hidden_states (float4 wide copy)
// ---------------------------------------------------------------------------
__global__ __launch_bounds__(256) void gather_kernel(const float* __restrict__ hidden,
                                                     const int* __restrict__ keep,
                                                     const int* __restrict__ dom,
                                                     float* __restrict__ out) {
    int b = blockIdx.x / 65, jj = blockIdx.x % 65;
    int k = dom[0]; k = k < 4 ? 4 : (k > 64 ? 64 : k);
    float4* d = (float4*)(out + ((size_t)(b * 65 + jj)) * D_);
    if (jj <= k) {
        int src = keep[b * 65 + jj];
        const float4* s = (const float4*)(hidden + ((size_t)(b * L_ + src)) * D_);
        d[threadIdx.x] = s[threadIdx.x];
    } else {
        d[threadIdx.x] = make_float4(0.f, 0.f, 0.f, 0.f);
    }
}

// ---------------------------------------------------------------------------
extern "C" void kernel_launch(void* const* d_in, const int* in_sizes, int n_in,
                              void* d_out, int out_size, void* d_ws, size_t ws_size,
                              hipStream_t stream) {
    const float* attn   = (const float*)d_in[0];
    const float* keys   = (const float*)d_in[1];
    const float* hidden = (const float*)d_in[2];
    const int*   dom    = (const int*)d_in[3];

    float* out = (float*)d_out;
    char*  ws  = (char*)d_ws;

    // workspace layout
    unsigned short* zb = (unsigned short*)ws;                      // 8*576*1024 bf16 = 9,437,184 B
    float* sattn = (float*)(ws + (size_t)B_ * N_ * D_ * 2);        // 4608 f32
    float* Hfeat = sattn + B_ * N_;                                // 4608 f32
    float* Isim  = Hfeat + B_ * N_;                                // 4608 f32
    int*   keep  = (int*)(Isim + B_ * N_);                         // 8*65 int

    float* dominant = out;                                         // [8,65,1024]
    float* scores   = out + (size_t)B_ * 65 * D_;                  // [8,576]

    attn_mean_kernel<<<(B_ * N_ + 255) / 256, 256, 0, stream>>>(attn, sattn);
    prep_kernel<<<B_ * N_, 256, 0, stream>>>(keys, zb, Hfeat);
    sim_entropy_kernel<<<B_ * (N_ / 16), 128, 0, stream>>>(zb, Isim);
    score_kernel<<<B_, 256, 0, stream>>>(sattn, Hfeat, Isim, scores);
    topk_kernel<<<B_, 256, 0, stream>>>(scores, dom, keep);
    gather_kernel<<<B_ * 65, 256, 0, stream>>>(hidden, keep, dom, dominant);
}
